// SimpleConvLSTM_4432406249575
// MI455X (gfx1250) — matrix operand order repaired and checked by reading, not compiled, not run on hardware
//
#include <hip/hip_runtime.h>
#include <stdint.h>

// ---------------------------------------------------------------------------
// Bidirectional 2-layer ConvLSTM for MI455X (gfx1250), bf16 WMMA pipeline.
// v_wmma_f32_16x16x32_bf16 with f32 accumulation. Per-timestep kernel fuses
// im2col(input)+im2col(hidden) GEMM + LSTM gates. A-operands are staged into
// a zero-padded LDS halo (one 3x18xC tile per block) so the hot K-loop is
// pure ds_load_b128 + global_load_b128(weights) + wmma, no predication.
// ---------------------------------------------------------------------------

#define T_STEPS 28
#define POS     4096        // B*H*W = 16*16*16
#define CO      256         // 4*F gate channels

typedef __attribute__((ext_vector_type(16))) __bf16 v16bf;
typedef __attribute__((ext_vector_type(8)))  float  v8f;

struct alignas(16) U4 { uint32_t a, b, c, d; };
union ABFrag { U4 q[2]; v16bf v; };

__device__ __forceinline__ uint16_t f2bf(float f) {
  uint32_t u = __float_as_uint(f);
  return (uint16_t)((u + 0x7FFFu + ((u >> 16) & 1u)) >> 16);   // RNE
}
__device__ __forceinline__ float bf2f(uint16_t h) {
  return __uint_as_float(((uint32_t)h) << 16);
}
__device__ __forceinline__ float sigmoidf(float x) {
  return 1.f / (1.f + __expf(-x));
}

// ---------------------------------------------------------------------------
__global__ void fill_zero(float* __restrict__ p, int n) {
  int i = blockIdx.x * blockDim.x + threadIdx.x;
  if (i < n) p[i] = 0.f;
}

// x [B,112,16,16,6] f32  ->  X0 [T][B*256][24] bf16 (time-major)
__global__ void embed_kernel(const float* __restrict__ xin, uint16_t* __restrict__ x0) {
  int idx = blockIdx.x * blockDim.x + threadIdx.x;   // 28*4096*24 exact
  int c  = idx % 24;
  int pp = idx / 24;                                 // t*4096 + b*256 + y*16 + x
  int xo = pp & 15, yo = (pp >> 4) & 15, b = (pp >> 8) & 15, t = pp >> 12;
  int lin = t * 24 + c;
  int tin = lin / 6, cin = lin % 6;
  x0[idx] = f2bf(xin[((((size_t)b * 112 + tin) * 16 + yo) * 16 + xo) * 6 + cin]);
}

// Pre-swizzle f32 weights [3][3][C][256] into bf16 WMMA B-fragment order with
// per-tap channel padding Cpad (multiple of 32). Frag layout (dwords):
//   frag[(((nt*NC + kc)*32 + lane)*8 + v)] = pack(bf16 K=k0, bf16 K=k0+1)
//   lane<16: k0 = kc*32 + 2v ; lane>=16: k0 = kc*32 + 16 + 2v ; n = nt*16+lane%16
__global__ void wprep(const float* __restrict__ w, uint32_t* __restrict__ frag,
                      int C, int Cpad, int NC) {
  int idx = blockIdx.x * blockDim.x + threadIdx.x;
  int total = 16 * NC * 256;
  if (idx >= total) return;
  int v    = idx & 7;
  int lane = (idx >> 3) & 31;
  int rest = idx >> 8;
  int kc   = rest % NC;
  int nt   = rest / NC;
  int n    = nt * 16 + (lane & 15);
  int hi   = lane >> 4;
  int kin  = kc * 32 + hi * 16 + 2 * v;      // padded K index
  int tap  = kin / Cpad;                     // chunk never straddles a tap
  int c0   = kin - tap * Cpad;
  float w0 = (c0     < C) ? w[((size_t)tap * C + c0    ) * 256 + n] : 0.f;
  float w1 = (c0 + 1 < C) ? w[((size_t)tap * C + c0 + 1) * 256 + n] : 0.f;
  frag[idx] = ((uint32_t)f2bf(w1) << 16) | (uint32_t)f2bf(w0);
}

// ---------------------------------------------------------------------------
// Stage the 3x18 spatial halo (rows yo-1..yo+1, cols -1..16) of one image
// into LDS, zero-padded spatially and in channels (C -> CPAD). Row stride is
// CPAD+8 halfs (16B skew) so the K-loop ds_load_b128 is bank-conflict free.
// ---------------------------------------------------------------------------
template<int C, int CPAD>
__device__ __forceinline__ void stage_halo(uint16_t* halo, const uint16_t* __restrict__ src,
                                           int b, int yo, int tid) {
  constexpr int NG = CPAD / 8;               // 16-byte groups per position
  constexpr int HS = CPAD + 8;               // padded row stride (halfs)
  for (int i = tid; i < 54 * NG; i += 128) {
    const int pos = i / NG;                  // 0..53 (NG is a power of two)
    const int cg  = i - pos * NG;
    const int c0  = cg * 8;
    const int r   = pos / 18;                // 0..2
    const int col = pos - r * 18;            // 0..17
    const int yy  = yo + r - 1;
    const int xx  = col - 1;
    U4 val = U4{0u, 0u, 0u, 0u};
    if (((unsigned)yy < 16u) && ((unsigned)xx < 16u) && (c0 < C))
      val = *(const U4*)(src + (size_t)((b << 8) + (yy << 4) + xx) * C + c0);
    *(U4*)(halo + pos * HS + c0) = val;
  }
}

// ---------------------------------------------------------------------------
// Conv-as-GEMM from the LDS halo: fully unrolled, zero predication.
// ---------------------------------------------------------------------------
template<int CPAD, int NCW>
__device__ __forceinline__ void conv_gemm_lds(const uint16_t* halo,
                                              const uint32_t* __restrict__ wf,
                                              int xo, int hi, int lane, int wave,
                                              v8f acc[4]) {
  constexpr int CPT = CPAD / 32;             // chunks per 3x3 tap
  constexpr int HS  = CPAD + 8;
  #pragma unroll
  for (int tap = 0; tap < 9; ++tap) {
    const int ky  = tap / 3;
    const int kx  = tap - ky * 3;
    const int pos = ky * 18 + xo + kx;       // halo position, 0..53
    const uint16_t* rowp = halo + pos * HS;
    #pragma unroll
    for (int ch = 0; ch < CPT; ++ch) {
      const int kcl = tap * CPT + ch;
      const int c0  = ch * 32 + hi * 16;     // K groups: c0..c0+15 (contiguous)
      ABFrag afr;
      afr.q[0] = *(const U4*)(rowp + c0);
      afr.q[1] = *(const U4*)(rowp + c0 + 8);
      #pragma unroll
      for (int j = 0; j < 4; ++j) {
        const int nt = (wave << 2) + j;
        const uint32_t* bp = wf + (((size_t)nt * NCW + kcl) * 32 + lane) * 8;
        ABFrag bfr;
        bfr.q[0] = *(const U4*)bp;
        bfr.q[1] = *(const U4*)(bp + 4);
        acc[j] = __builtin_amdgcn_wmma_f32_16x16x32_bf16(
            false, afr.v, false, bfr.v, (short)0, acc[j], false, false);
      }
    }
  }
}

// ---------------------------------------------------------------------------
// One recurrent timestep, one direction. Block = one image row (16 positions)
// x all 256 gate channels; 4 waves, each owning 4 N-tiles (= one gate).
// ---------------------------------------------------------------------------
template<int SRCC, int SRCCPAD, int NCX>
__global__ void __launch_bounds__(128)
lstm_step(const uint16_t* __restrict__ src,
          const uint32_t* __restrict__ wxFrag, const uint32_t* __restrict__ whFrag,
          const float* __restrict__ bias,
          const uint16_t* __restrict__ hread, uint16_t* __restrict__ hwrite,
          float* __restrict__ cstate,
          uint16_t* __restrict__ yout, int tt, int coff) {
  constexpr int HALO_C = (SRCCPAD > 64) ? SRCCPAD : 64;
  __shared__ uint16_t halo[54 * (HALO_C + 8)];       // reused for src then h
  __shared__ float zt[16][CO];                       // 16KB gate staging tile

  const int tid  = threadIdx.x;
  const int lane = tid & 31;
  const int wave = tid >> 5;
  const int m    = blockIdx.x;                       // M tile (0..255)
  const int b    = m >> 4;                           // block-uniform image / row
  const int yo   = m & 15;
  const int hi   = lane >> 4;
  const int rsel = lane & 15;
  const int xo   = rsel;                             // this lane's A-row column

  v8f zero = {0.f, 0.f, 0.f, 0.f, 0.f, 0.f, 0.f, 0.f};
  v8f acc[4];
  #pragma unroll
  for (int j = 0; j < 4; ++j) acc[j] = zero;

  // ---- input-to-hidden contribution (time tt) ----
  stage_halo<SRCC, SRCCPAD>(halo, src + (size_t)tt * POS * SRCC, b, yo, tid);
  __syncthreads();
  conv_gemm_lds<SRCCPAD, NCX>(halo, wxFrag, xo, hi, lane, wave, acc);
  __syncthreads();

  // ---- hidden-to-hidden contribution (h_{t-1}) ----
  stage_halo<64, 64>(halo, hread, b, yo, tid);
  __syncthreads();
  conv_gemm_lds<64, 18>(halo, whFrag, xo, hi, lane, wave, acc);

  // ---- bias + spill gate pre-activations to LDS (C layout: row=r+8*hi) ----
  #pragma unroll
  for (int j = 0; j < 4; ++j) {
    const int col = (wave << 6) + (j << 4) + rsel;
    const float bv = bias[col];
    #pragma unroll
    for (int r = 0; r < 8; ++r)
      zt[r + (hi << 3)][col] = acc[j][r] + bv;
  }
  __syncthreads();

  // ---- LSTM cell update: 16 rows x 64 cells, 8 cells/thread ----
  #pragma unroll
  for (int k = 0; k < 8; ++k) {
    const int cell = (tid << 3) + k;
    const int row  = cell >> 6;
    const int cc   = cell & 63;
    const float ig = sigmoidf(zt[row][cc]);
    const float fg = sigmoidf(zt[row][64 + cc]);
    const float gg = tanhf(zt[row][128 + cc]);
    const float og = sigmoidf(zt[row][192 + cc]);
    const size_t s = (size_t)((m << 4) + row) * 64 + cc;
    const float cn = fg * cstate[s] + ig * gg;
    cstate[s] = cn;
    const float h = og * tanhf(cn);
    const uint16_t hb = f2bf(h);
    hwrite[s] = hb;                                  // double-buffered recurrence
    yout[((size_t)tt * POS + ((m << 4) + row)) * 128 + coff + cc] = hb;
  }
}

// Final TimeDistributed 3x3 conv, 128 -> 1, f32 out [B][T][16][16]
__global__ void out_conv(const uint16_t* __restrict__ y1, const float* __restrict__ wout,
                         const float* __restrict__ bout, float* __restrict__ out) {
  int idx = blockIdx.x * blockDim.x + threadIdx.x;   // 114688 exact
  int xo = idx & 15, yo = (idx >> 4) & 15;
  int t = (idx >> 8) % T_STEPS;
  int b = idx / (256 * T_STEPS);
  float sum = bout[0];
  for (int tap = 0; tap < 9; ++tap) {
    int ky = tap / 3, kx = tap % 3;
    int yy = yo + ky - 1, xx = xo + kx - 1;
    if ((unsigned)yy >= 16u || (unsigned)xx >= 16u) continue;
    const uint16_t* row = y1 + ((size_t)t * POS + (b << 8) + (yy << 4) + xx) * 128;
    const float* wr = wout + tap * 128;
    #pragma unroll 8
    for (int c = 0; c < 128; ++c) sum += bf2f(row[c]) * wr[c];
  }
  out[idx] = sum;
}

// ---------------------------------------------------------------------------
extern "C" void kernel_launch(void* const* d_in, const int* in_sizes, int n_in,
                              void* d_out, int out_size, void* d_ws, size_t ws_size,
                              hipStream_t stream) {
  const float* xin  = (const float*)d_in[0];
  const float* wx0f = (const float*)d_in[1];
  const float* wh0f = (const float*)d_in[2];
  const float* b0f  = (const float*)d_in[3];
  const float* wx0b = (const float*)d_in[4];
  const float* wh0b = (const float*)d_in[5];
  const float* b0b  = (const float*)d_in[6];
  const float* wx1f = (const float*)d_in[7];
  const float* wh1f = (const float*)d_in[8];
  const float* b1f  = (const float*)d_in[9];
  const float* wx1b = (const float*)d_in[10];
  const float* wh1b = (const float*)d_in[11];
  const float* b1b  = (const float*)d_in[12];
  const float* wout = (const float*)d_in[13];
  const float* bout = (const float*)d_in[14];

  // -------- workspace carve (~72 MB total), 256B-aligned slabs --------
  char* ws = (char*)d_ws;
  size_t off = 0;
  auto carve = [&](size_t bytes) -> char* {
    char* p = ws + off;
    off += (bytes + 255) & ~(size_t)255;
    return p;
  };
  uint16_t* X0 = (uint16_t*)carve((size_t)T_STEPS * POS * 24 * 2);   // 5.25 MB
  uint16_t* Y0 = (uint16_t*)carve((size_t)T_STEPS * POS * 128 * 2);  // 28 MB
  uint16_t* Y1 = (uint16_t*)carve((size_t)T_STEPS * POS * 128 * 2);  // 28 MB
  uint32_t* fx0f = (uint32_t*)carve((size_t)36864 * 4);
  uint32_t* fx0b = (uint32_t*)carve((size_t)36864 * 4);
  uint32_t* fh0f = (uint32_t*)carve((size_t)73728 * 4);
  uint32_t* fh0b = (uint32_t*)carve((size_t)73728 * 4);
  uint32_t* fx1f = (uint32_t*)carve((size_t)147456 * 4);
  uint32_t* fx1b = (uint32_t*)carve((size_t)147456 * 4);
  uint32_t* fh1f = (uint32_t*)carve((size_t)73728 * 4);
  uint32_t* fh1b = (uint32_t*)carve((size_t)73728 * 4);
  // recurrent state: 4 chains x (c f32 1MB + two bf16 h buffers 0.5MB each)
  char* state = carve((size_t)4 * 2 * 1024 * 1024);
  float*    cst[4];
  uint16_t* hb[4][2];
  for (int i = 0; i < 4; ++i) {
    char* p = state + (size_t)i * 2 * 1024 * 1024;
    cst[i]   = (float*)p;
    hb[i][0] = (uint16_t*)(p + 1024 * 1024);
    hb[i][1] = (uint16_t*)(p + 1024 * 1024 + 512 * 1024);
  }

  // -------- prologue: zero state, embed input, swizzle weights --------
  fill_zero<<<8192, 256, 0, stream>>>((float*)state, 4 * 2 * 1024 * 1024 / 4);
  embed_kernel<<<10752, 256, 0, stream>>>(xin, X0);
  wprep<<<144, 256, 0, stream>>>(wx0f, fx0f, 24, 32, 9);
  wprep<<<144, 256, 0, stream>>>(wx0b, fx0b, 24, 32, 9);
  wprep<<<288, 256, 0, stream>>>(wh0f, fh0f, 64, 64, 18);
  wprep<<<288, 256, 0, stream>>>(wh0b, fh0b, 64, 64, 18);
  wprep<<<576, 256, 0, stream>>>(wx1f, fx1f, 128, 128, 36);
  wprep<<<576, 256, 0, stream>>>(wx1b, fx1b, 128, 128, 36);
  wprep<<<288, 256, 0, stream>>>(wh1f, fh1f, 64, 64, 18);
  wprep<<<288, 256, 0, stream>>>(wh1b, fh1b, 64, 64, 18);

  // -------- layer 0: fwd scans t ascending, bwd reads/writes reversed time --------
  for (int s = 0; s < T_STEPS; ++s) {
    int rb = s & 1, wb = rb ^ 1;
    lstm_step<24, 32, 9><<<256, 128, 0, stream>>>(X0, fx0f, fh0f, b0f,
        hb[0][rb], hb[0][wb], cst[0], Y0, s, 0);
    lstm_step<24, 32, 9><<<256, 128, 0, stream>>>(X0, fx0b, fh0b, b0b,
        hb[1][rb], hb[1][wb], cst[1], Y0, T_STEPS - 1 - s, 64);
  }
  // -------- layer 1 (input = concat Y0, C=128) --------
  for (int s = 0; s < T_STEPS; ++s) {
    int rb = s & 1, wb = rb ^ 1;
    lstm_step<128, 128, 36><<<256, 128, 0, stream>>>(Y0, fx1f, fh1f, b1f,
        hb[2][rb], hb[2][wb], cst[2], Y1, s, 0);
    lstm_step<128, 128, 36><<<256, 128, 0, stream>>>(Y0, fx1b, fh1b, b1b,
        hb[3][rb], hb[3][wb], cst[3], Y1, T_STEPS - 1 - s, 64);
  }
  // -------- final conv --------
  out_conv<<<448, 256, 0, stream>>>(Y1, wout, bout, (float*)d_out);

  (void)in_sizes; (void)n_in; (void)out_size; (void)ws_size;
}
